// WindowAttention_22419729285946
// MI455X (gfx1250) — matrix operand order repaired and checked by reading, not compiled, not run on hardware
//
#include <hip/hip_runtime.h>
#include <hip/hip_bf16.h>

typedef __attribute__((ext_vector_type(16))) _Float16 v16h;
typedef __attribute__((ext_vector_type(8)))  _Float16 v8h;
typedef __attribute__((ext_vector_type(8)))  float    v8f;

#define IMG    96
#define CCH    256
#define NTOK   64      // 8x8 tokens per window
#define NWIN   23      // window origins per axis
#define NHEADS 8
#define HDIM   32

// ---------- WMMA fragment loaders (layouts per CDNA5 ISA 7.12.2) ----------

// A 16x32 f16: lane (0-15)->row M, halves: e0..7 -> K = kbase+8*(lane>>4)+e,
//              e8..15 -> K = kbase+16+8*(lane>>4)+e
__device__ __forceinline__ v16h load_a16(const _Float16* base, int ld,
                                         int mbase, int kbase, int lane) {
    const _Float16* p = base + (mbase + (lane & 15)) * ld + kbase + ((lane >> 4) << 3);
    v8h lo = *(const v8h*)(p);
    v8h hi = *(const v8h*)(p + 16);
    return __builtin_shufflevector(lo, hi, 0,1,2,3,4,5,6,7,8,9,10,11,12,13,14,15);
}

// B 32x16 f16: lane (0-15)->col N, halves: e0..7 -> K = kbase+8*(lane>>4)+e,
//              e8..15 -> K = kbase+16+8*(lane>>4)+e
__device__ __forceinline__ v16h load_b16(const _Float16* base, int ld,
                                         int nbase, int kbase, int lane) {
    const _Float16* p = base + (nbase + (lane & 15)) * ld + kbase + ((lane >> 4) << 3);
    v8h lo = *(const v8h*)(p);
    v8h hi = *(const v8h*)(p + 16);
    return __builtin_shufflevector(lo, hi, 0,1,2,3,4,5,6,7,8,9,10,11,12,13,14,15);
}

__device__ __forceinline__ v8f wmma16(v16h a, v16h b, v8f c) {
    return __builtin_amdgcn_wmma_f32_16x16x32_f16(false, a, false, b, (short)0, c,
                                                  false, false);
}

// ---------- prep: f32->f16 weights, expand relative-position bias ----------

__global__ void prep_kernel(const float* __restrict__ Wqkv,
                            const float* __restrict__ Wproj,
                            const float* __restrict__ relTab,
                            _Float16* __restrict__ Whqkv,
                            _Float16* __restrict__ Whproj,
                            float* __restrict__ biasg) {
    int i = blockIdx.x * blockDim.x + threadIdx.x;
    if (i < 768 * 256) Whqkv[i]  = (_Float16)Wqkv[i];
    if (i < 256 * 256) Whproj[i] = (_Float16)Wproj[i];
    if (i < NHEADS * NTOK * NTOK) {
        int h = i >> 12;
        int n = (i >> 6) & 63;
        int m = i & 63;
        int di = (n >> 3) - (m >> 3) + 7;
        int dj = (n & 7)  - (m & 7)  + 7;
        biasg[i] = relTab[(di * 15 + dj) * NHEADS + h];
    }
}

// ---------- main fused window-attention kernel: 1 block = 1 window ----------

__global__ __launch_bounds__(256, 1) void win_attn_kernel(
    const float* __restrict__ vid,
    const float* __restrict__ modulator,
    const float* __restrict__ bqkv,
    const float* __restrict__ bproj,
    const _Float16* __restrict__ Whqkv,
    const _Float16* __restrict__ Whproj,
    const float* __restrict__ biasg,
    float* __restrict__ accb) {
    extern __shared__ _Float16 smem[];
    _Float16* xh = smem;               // 64x256 halves (reused as O after attn)
    _Float16* qh = xh + NTOK * CCH;    // 64x256 (Q, pre-scaled)
    _Float16* kh = qh + NTOK * CCH;    // 64x256 (K, row-major: contiguous in d)
    _Float16* vT = kh + NTOK * CCH;    // 256x64 (V transposed: contiguous in m)
    _Float16* pr = vT + NTOK * CCH;    // 8x64x64 softmax probs

    const int b    = blockIdx.x;
    const int t    = b / (NWIN * NWIN);
    const int rem  = b % (NWIN * NWIN);
    const int ry   = (rem / NWIN) * 4;
    const int cx   = (rem % NWIN) * 4;

    const int tid  = threadIdx.x;
    const int lane = tid & 31;
    const int wv   = tid >> 5;            // 8 waves
    const int lo16 = lane & 15;
    const int rb   = (lane >> 4) << 3;    // C/D row offset of the lane-half

    // ---- Phase 1: gather window + modulator -> f16 x in LDS ----
    for (int i = tid; i < NTOK * CCH / 4; i += 256) {
        int n  = i >> 6;
        int c4 = (i & 63) << 2;
        int py = n >> 3, px = n & 7;
        const float4 vv = *(const float4*)(vid + (((t * IMG + ry + py) * IMG) + cx + px) * CCH + c4);
        const float4 mm = *(const float4*)(modulator + n * CCH + c4);
        _Float16* d = xh + n * CCH + c4;
        d[0] = (_Float16)(vv.x + mm.x);
        d[1] = (_Float16)(vv.y + mm.y);
        d[2] = (_Float16)(vv.z + mm.z);
        d[3] = (_Float16)(vv.w + mm.w);
    }
    __syncthreads();

    // ---- Phase 2: QKV = x @ Wqkv^T + bqkv  (M=64, N=768, K=256) ----
    const float scale = 0.17677669529663687f;  // 1/sqrt(32)
    for (int jt = wv * 6; jt < wv * 6 + 6; ++jt) {   // 48 N-tiles / 8 waves
        v8f c0 = {}, c1 = {}, c2 = {}, c3 = {};
        #pragma unroll
        for (int ks = 0; ks < 8; ++ks) {
            v16h bf = load_b16(Whqkv, CCH, jt * 16, ks * 32, lane);
            c0 = wmma16(load_a16(xh, CCH,  0, ks * 32, lane), bf, c0);
            c1 = wmma16(load_a16(xh, CCH, 16, ks * 32, lane), bf, c1);
            c2 = wmma16(load_a16(xh, CCH, 32, ks * 32, lane), bf, c2);
            c3 = wmma16(load_a16(xh, CCH, 48, ks * 32, lane), bf, c3);
        }
        const int col   = jt * 16 + lo16;   // 0..767
        const int which = col >> 8;         // 0=q 1=k 2=v
        const int cc    = col & 255;
        const float bv  = bqkv[col];
        v8f A[4] = {c0, c1, c2, c3};
        #pragma unroll
        for (int mt = 0; mt < 4; ++mt)
            #pragma unroll
            for (int r = 0; r < 8; ++r) {
                const int n = mt * 16 + rb + r;
                const float val = A[mt][r] + bv;
                if (which == 0)      qh[n * CCH + cc] = (_Float16)(val * scale);
                else if (which == 1) kh[n * CCH + cc] = (_Float16)val;
                else                 vT[cc * NTOK + n] = (_Float16)val;
            }
    }
    __syncthreads();

    // ---- Phase 3: per-head attention scores + softmax (head h = wave id) ----
    const int h = wv;
    v8f S[4][4];
    #pragma unroll
    for (int mt = 0; mt < 4; ++mt) {
        v16h aq = load_a16(qh, CCH, mt * 16, h * HDIM, lane);
        #pragma unroll
        for (int nt = 0; nt < 4; ++nt) {
            v8f z = {};
            S[mt][nt] = wmma16(aq, load_b16(kh, CCH, nt * 16, h * HDIM, lane), z);
        }
    }
    const float*    bh = biasg + h * NTOK * NTOK;
    _Float16*       ph = pr    + h * NTOK * NTOK;
    #pragma unroll
    for (int mt = 0; mt < 4; ++mt)
        #pragma unroll
        for (int r = 0; r < 8; ++r) {
            const int n = mt * 16 + rb + r;
            float s0 = S[mt][0][r] + bh[n * 64 +  0 + lo16];
            float s1 = S[mt][1][r] + bh[n * 64 + 16 + lo16];
            float s2 = S[mt][2][r] + bh[n * 64 + 32 + lo16];
            float s3 = S[mt][3][r] + bh[n * 64 + 48 + lo16];
            float mx = fmaxf(fmaxf(s0, s1), fmaxf(s2, s3));
            #pragma unroll
            for (int o = 8; o >= 1; o >>= 1) mx = fmaxf(mx, __shfl_xor(mx, o, 16));
            s0 = __expf(s0 - mx); s1 = __expf(s1 - mx);
            s2 = __expf(s2 - mx); s3 = __expf(s3 - mx);
            float sm = s0 + s1 + s2 + s3;
            #pragma unroll
            for (int o = 8; o >= 1; o >>= 1) sm += __shfl_xor(sm, o, 16);
            const float inv = 1.0f / sm;
            ph[n * 64 +  0 + lo16] = (_Float16)(s0 * inv);
            ph[n * 64 + 16 + lo16] = (_Float16)(s1 * inv);
            ph[n * 64 + 32 + lo16] = (_Float16)(s2 * inv);
            ph[n * 64 + 48 + lo16] = (_Float16)(s3 * inv);
        }
    __syncthreads();

    // ---- Phase 4: O_h = P @ V_h  (M=64, N=32, K=64), O reuses x buffer ----
    _Float16* oh = xh;
    #pragma unroll
    for (int mt = 0; mt < 4; ++mt)
        #pragma unroll
        for (int vt = 0; vt < 2; ++vt) {
            v8f acc = {};
            #pragma unroll
            for (int ks = 0; ks < 2; ++ks)
                acc = wmma16(load_a16(ph, 64, mt * 16, ks * 32, lane),
                             load_b16(vT, NTOK, h * HDIM + vt * 16, ks * 32, lane), acc);
            const int ccol = h * HDIM + vt * 16 + lo16;
            #pragma unroll
            for (int r = 0; r < 8; ++r)
                oh[(mt * 16 + rb + r) * CCH + ccol] = (_Float16)acc[r];
        }
    __syncthreads();

    // ---- Phase 5: proj = O @ Wproj^T + bproj, overlap-add via atomics ----
    #pragma unroll
    for (int jt2 = 0; jt2 < 2; ++jt2) {
        const int jt = wv * 2 + jt2;     // 16 N-tiles / 8 waves
        v8f c0 = {}, c1 = {}, c2 = {}, c3 = {};
        #pragma unroll
        for (int ks = 0; ks < 8; ++ks) {
            v16h bf = load_b16(Whproj, CCH, jt * 16, ks * 32, lane);
            c0 = wmma16(load_a16(oh, CCH,  0, ks * 32, lane), bf, c0);
            c1 = wmma16(load_a16(oh, CCH, 16, ks * 32, lane), bf, c1);
            c2 = wmma16(load_a16(oh, CCH, 32, ks * 32, lane), bf, c2);
            c3 = wmma16(load_a16(oh, CCH, 48, ks * 32, lane), bf, c3);
        }
        const int ccol = jt * 16 + lo16;
        const float bp = bproj[ccol];
        v8f A[4] = {c0, c1, c2, c3};
        #pragma unroll
        for (int mt = 0; mt < 4; ++mt)
            #pragma unroll
            for (int r = 0; r < 8; ++r) {
                const int n  = mt * 16 + rb + r;
                const int py = n >> 3, px = n & 7;
                atomicAdd(accb + (((t * IMG + ry + py) * IMG) + cx + px) * CCH + ccol,
                          A[mt][r] + bp);
            }
    }
}

// ---------- normalize by analytic overlap count ----------

__device__ __forceinline__ float cover1(int v) {
    int hi = v >> 2; if (hi > 22) hi = 22;      // floor(v/4) clamped
    int lo = v - 7;  lo = (lo > 0) ? ((lo + 3) >> 2) : 0;  // ceil((v-7)/4) clamped
    return (float)(hi - lo + 1);
}

__global__ void norm_kernel(const float* __restrict__ accb, float* __restrict__ out) {
    int i = blockIdx.x * blockDim.x + threadIdx.x;
    if (i >= 4 * IMG * IMG * CCH) return;
    int rest = i >> 8;
    int x = rest % IMG;
    int y = (rest / IMG) % IMG;
    out[i] = accb[i] / (cover1(y) * cover1(x) + 1e-10f);
}

// ---------- launch ----------

extern "C" void kernel_launch(void* const* d_in, const int* in_sizes, int n_in,
                              void* d_out, int out_size, void* d_ws, size_t ws_size,
                              hipStream_t stream) {
    const float* vid   = (const float*)d_in[0];
    const float* mod   = (const float*)d_in[1];
    const float* rpt   = (const float*)d_in[2];
    const float* Wqkv  = (const float*)d_in[3];
    const float* bqkv  = (const float*)d_in[4];
    const float* Wproj = (const float*)d_in[5];
    const float* bproj = (const float*)d_in[6];
    float* out = (float*)d_out;

    char* ws = (char*)d_ws;
    _Float16* Whqkv  = (_Float16*)(ws);                 // 768*256*2   = 393216 B
    _Float16* Whproj = (_Float16*)(ws + 393216);        // 256*256*2   = 131072 B
    float*    biasg  = (float*)   (ws + 524288);        // 8*64*64*4   = 131072 B
    float*    accb   = (float*)   (ws + 655360);        // 4*96*96*256*4
    const size_t accBytes = (size_t)4 * IMG * IMG * CCH * sizeof(float);

    hipMemsetAsync(accb, 0, accBytes, stream);
    prep_kernel<<<768, 256, 0, stream>>>(Wqkv, Wproj, rpt, Whqkv, Whproj, biasg);

    const int smemBytes = (4 * NTOK * CCH + NHEADS * NTOK * NTOK) * (int)sizeof(_Float16); // 196608
    hipFuncSetAttribute(reinterpret_cast<const void*>(win_attn_kernel),
                        hipFuncAttributeMaxDynamicSharedMemorySize, smemBytes);
    win_attn_kernel<<<4 * NWIN * NWIN, 256, smemBytes, stream>>>(
        vid, mod, bqkv, bproj, Whqkv, Whproj, biasg, accb);

    const int total = 4 * IMG * IMG * CCH;
    norm_kernel<<<(total + 255) / 256, 256, 0, stream>>>(accb, out);
}
